// NetA_NodeOnly_75840532513222
// MI455X (gfx1250) — compile-verified, matrix operand order checked by reading, not run on hardware
//
#include <hip/hip_runtime.h>
#include <hip/hip_bf16.h>
#include <stdint.h>

#define N_NODES 20000
#define N_EDGES 320000
#define N_GRAPH 64
#define QSTRIDE 1024   // fused Q|K|V|S columns

typedef __attribute__((ext_vector_type(16))) __bf16 v16bf;
typedef __attribute__((ext_vector_type(8)))  float  v8f;

union FragBF { unsigned u[8]; v16bf v; };

// ---------- helpers ----------
__device__ __forceinline__ unsigned pk_bf16(float lo, float hi) {
    unsigned a = __float_as_uint(lo); a += 0x7FFFu + ((a >> 16) & 1u);
    unsigned b = __float_as_uint(hi); b += 0x7FFFu + ((b >> 16) & 1u);
    return (a >> 16) | (b & 0xFFFF0000u);
}
__device__ __forceinline__ unsigned short bf16rnd(float x) {
    unsigned u = __float_as_uint(x); u += 0x7FFFu + ((u >> 16) & 1u);
    return (unsigned short)(u >> 16);
}

// ---------- weight repack: Wq|Wk|Wv|Ws (each [256,256] f32) -> WcatT [1024][256] bf16 ----------
__global__ __launch_bounds__(256) void conv_weights(
    const float* __restrict__ Wq, const float* __restrict__ Wk,
    const float* __restrict__ Wv, const float* __restrict__ Ws,
    const float* __restrict__ bq, const float* __restrict__ bk,
    const float* __restrict__ bv, const float* __restrict__ bs,
    unsigned short* __restrict__ Wt, float* __restrict__ bcat)
{
    int id = blockIdx.x * 256 + threadIdx.x;     // j*256 + k
    if (id >= 1024 * 256) return;
    int j = id >> 8, k = id & 255;
    int sel = j >> 8, jj = j & 255;
    const float* W = (sel == 0) ? Wq : (sel == 1) ? Wk : (sel == 2) ? Wv : Ws;
    Wt[(size_t)j * 256 + k] = bf16rnd(W[k * 256 + jj]);   // transpose: [out][k]
    if (k == 0) {
        const float* bb = (sel == 0) ? bq : (sel == 1) ? bk : (sel == 2) ? bv : bs;
        bcat[j] = bb[jj];
    }
}

// ---------- embedding gather: h0[n][f*32+c] = emb[x[n][f]][c] ----------
__global__ __launch_bounds__(256) void embed_gather(
    const int* __restrict__ x, const float* __restrict__ emb, float* __restrict__ h0)
{
    int id = blockIdx.x * 256 + threadIdx.x;     // n*256 + c
    int n = id >> 8, c = id & 255;
    int f = c >> 5, cc = c & 31;
    int tok = x[n * 8 + f];
    h0[id] = emb[(size_t)tok * 32 + cc];
}

// ---------- CSR build (dst-indexed), once per launch ----------
__global__ __launch_bounds__(256) void csr_count(const int* __restrict__ ei, int* __restrict__ counts)
{
    int e = blockIdx.x * 256 + threadIdx.x;
    if (e >= N_EDGES) return;
    atomicAdd(&counts[ei[N_EDGES + e]], 1);
}

__global__ __launch_bounds__(256) void csr_scan(
    const int* __restrict__ counts, int* __restrict__ offsets, int* __restrict__ cursor)
{
    __shared__ int part[256];
    __shared__ int total;
    const int CH = (N_NODES + 255) / 256;   // 79
    int t = threadIdx.x;
    int base = t * CH;
    int s = 0;
    for (int i = 0; i < CH; i++) { int idx = base + i; if (idx < N_NODES) s += counts[idx]; }
    part[t] = s;
    __syncthreads();
    if (t == 0) {
        int run = 0;
        for (int i = 0; i < 256; i++) { int v = part[i]; part[i] = run; run += v; }
        total = run;
    }
    __syncthreads();
    int run = part[t];
    for (int i = 0; i < CH; i++) {
        int idx = base + i;
        if (idx < N_NODES) { offsets[idx] = run; cursor[idx] = run; run += counts[idx]; }
    }
    if (t == 0) offsets[N_NODES] = total;
}

__global__ __launch_bounds__(256) void csr_fill(
    const int* __restrict__ ei, int* __restrict__ cursor, int* __restrict__ eids)
{
    int e = blockIdx.x * 256 + threadIdx.x;
    if (e >= N_EDGES) return;
    int p = atomicAdd(&cursor[ei[N_EDGES + e]], 1);
    eids[p] = e;
}

// ---------- fused QKVS WMMA GEMM: Y[N,1024] = bf16(X[N,256]) @ WcatT^T + bcat ----------
__global__ __launch_bounds__(256) void wmma_gemm_qkvs(
    const float* __restrict__ X, const unsigned short* __restrict__ Wt,
    const float* __restrict__ bcat, float* __restrict__ Y, int M, int K)
{
    __shared__ unsigned lds_a[128 * 17];   // 128 rows x 16 uints (32 bf16) + pad
    __shared__ unsigned lds_b[128 * 17];

    const int t    = threadIdx.x;
    const int lane = t & 31;
    const int w    = t >> 5;
    const int wm   = w >> 1;   // 0..3 -> 32 rows each
    const int wn   = w & 1;    // 0..1 -> 64 cols each
    const int fi   = lane & 15;
    const int fh   = lane >> 4;
    const int bm   = blockIdx.x * 128;
    const int bn   = blockIdx.y * 128;

    v8f acc[2][4];
#pragma unroll
    for (int i = 0; i < 2; i++)
#pragma unroll
        for (int j = 0; j < 4; j++)
#pragma unroll
            for (int r = 0; r < 8; r++) acc[i][j][r] = 0.0f;

    const int lr = t >> 1;   // 0..127 (tile row)
    const int lh = t & 1;    // which 16-wide k half

    for (int k0 = 0; k0 < K; k0 += 32) {
        {   // stage A (f32 -> bf16 packed pairs)
            int gr = bm + lr; if (gr >= M) gr = M - 1;
            const float4* s = (const float4*)(X + (size_t)gr * K + k0 + lh * 16);
            float4 f0 = s[0], f1 = s[1], f2 = s[2], f3 = s[3];
            unsigned* d = lds_a + lr * 17 + lh * 8;
            d[0] = pk_bf16(f0.x, f0.y); d[1] = pk_bf16(f0.z, f0.w);
            d[2] = pk_bf16(f1.x, f1.y); d[3] = pk_bf16(f1.z, f1.w);
            d[4] = pk_bf16(f2.x, f2.y); d[5] = pk_bf16(f2.z, f2.w);
            d[6] = pk_bf16(f3.x, f3.y); d[7] = pk_bf16(f3.z, f3.w);
        }
        {   // stage B (already bf16, [out][k] so same fragment layout as A)
            const uint4* s = (const uint4*)(Wt + (size_t)(bn + lr) * K + k0 + lh * 16);
            uint4 q0 = s[0], q1 = s[1];
            unsigned* d = lds_b + lr * 17 + lh * 8;
            d[0] = q0.x; d[1] = q0.y; d[2] = q0.z; d[3] = q0.w;
            d[4] = q1.x; d[5] = q1.y; d[6] = q1.z; d[7] = q1.w;
        }
        __syncthreads();

        FragBF a[2], b[4];
#pragma unroll
        for (int i = 0; i < 2; i++) {
            const unsigned* p = lds_a + (wm * 32 + i * 16 + fi) * 17 + fh * 4;
            a[i].u[0] = p[0]; a[i].u[1] = p[1]; a[i].u[2] = p[2];  a[i].u[3] = p[3];
            a[i].u[4] = p[8]; a[i].u[5] = p[9]; a[i].u[6] = p[10]; a[i].u[7] = p[11];
        }
#pragma unroll
        for (int j = 0; j < 4; j++) {
            const unsigned* p = lds_b + (wn * 64 + j * 16 + fi) * 17 + fh * 4;
            b[j].u[0] = p[0]; b[j].u[1] = p[1]; b[j].u[2] = p[2];  b[j].u[3] = p[3];
            b[j].u[4] = p[8]; b[j].u[5] = p[9]; b[j].u[6] = p[10]; b[j].u[7] = p[11];
        }
#pragma unroll
        for (int i = 0; i < 2; i++)
#pragma unroll
            for (int j = 0; j < 4; j++)
                acc[i][j] = __builtin_amdgcn_wmma_f32_16x16x32_bf16(
                    false, a[i].v, false, b[j].v, (short)0, acc[i][j], false, false);
        __syncthreads();
    }

    const bool full = (bm + 128 <= M);
#pragma unroll
    for (int i = 0; i < 2; i++)
#pragma unroll
        for (int j = 0; j < 4; j++) {
            int n = bn + wn * 64 + j * 16 + fi;
            float bv = bcat[n];
            int mbase = bm + wm * 32 + i * 16 + 8 * fh;
            float* yp = Y + (size_t)mbase * QSTRIDE + n;
            if (full) {
#pragma unroll
                for (int r = 0; r < 8; r++) yp[(size_t)r * QSTRIDE] = acc[i][j][r] + bv;
            } else {
#pragma unroll
                for (int r = 0; r < 8; r++)
                    if (mbase + r < M) yp[(size_t)r * QSTRIDE] = acc[i][j][r] + bv;
            }
        }
}

// ---------- edge pass: logits only (one wave per edge, no atomics) ----------
__global__ __launch_bounds__(256) void edge_logits(
    const int* __restrict__ ei, const float* __restrict__ ea,
    const float* __restrict__ We, const float* __restrict__ Qcat,
    float* __restrict__ logits)
{
    __shared__ float sWe[6 * 256];
    for (int i = threadIdx.x; i < 6 * 256; i += 256) sWe[i] = We[i];
    __syncthreads();

    int wv = threadIdx.x >> 5, lane = threadIdx.x & 31;
    int e = blockIdx.x * 8 + wv;
    if (e >= N_EDGES) return;
    int src = ei[e], dst = ei[N_EDGES + e];

    float a0 = ea[e * 6 + 0], a1 = ea[e * 6 + 1], a2 = ea[e * 6 + 2];
    float a3 = ea[e * 6 + 3], a4 = ea[e * 6 + 4], a5 = ea[e * 6 + 5];

    const float* q  = Qcat + (size_t)dst * QSTRIDE;           // Q at col 0
    const float* kk = Qcat + (size_t)src * QSTRIDE + 256;     // K at col 256
    float part = 0.0f;
#pragma unroll
    for (int i = 0; i < 8; i++) {
        int c = i * 32 + lane;
        float ev = a0 * sWe[c] + a1 * sWe[256 + c] + a2 * sWe[512 + c]
                 + a3 * sWe[768 + c] + a4 * sWe[1024 + c] + a5 * sWe[1280 + c];
        part += q[c] * (kk[c] + ev);
    }
#pragma unroll
    for (int o = 16; o > 0; o >>= 1) part += __shfl_down(part, o, 32);
    if (lane == 0) logits[e] = part * 0.0625f;   // 1/sqrt(256)
}

// ---------- gather aggregation: one wave per dst node; fused softmax + relu + skip ----------
__global__ __launch_bounds__(256) void node_aggregate(
    const int* __restrict__ ei, const float* __restrict__ ea,
    const float* __restrict__ We, const float* __restrict__ Qcat,
    const float* __restrict__ logits,
    const int* __restrict__ offsets, const int* __restrict__ eids,
    float* __restrict__ nf)
{
    __shared__ float sWe[6 * 256];
    for (int i = threadIdx.x; i < 6 * 256; i += 256) sWe[i] = We[i];
    __syncthreads();

    int wv = threadIdx.x >> 5, lane = threadIdx.x & 31;
    int n = blockIdx.x * 8 + wv;
    if (n >= N_NODES) return;

    int beg = offsets[n], end = offsets[n + 1];
    float acc[8];
#pragma unroll
    for (int i = 0; i < 8; i++) acc[i] = 0.0f;
    float den = 0.0f;

    if (beg < end) {
        float m = -3.402823466e38f;
        for (int p = beg; p < end; p++) m = fmaxf(m, logits[eids[p]]);
        for (int p = beg; p < end; p++) {
            int e = eids[p];
            float wgt = __expf(logits[e] - m);
            den += wgt;
            int src = ei[e];
            float a0 = ea[e * 6 + 0], a1 = ea[e * 6 + 1], a2 = ea[e * 6 + 2];
            float a3 = ea[e * 6 + 3], a4 = ea[e * 6 + 4], a5 = ea[e * 6 + 5];
            const float* vv = Qcat + (size_t)src * QSTRIDE + 512;   // V at col 512
#pragma unroll
            for (int i = 0; i < 8; i++) {
                int c = i * 32 + lane;
                float ev = a0 * sWe[c] + a1 * sWe[256 + c] + a2 * sWe[512 + c]
                         + a3 * sWe[768 + c] + a4 * sWe[1024 + c] + a5 * sWe[1280 + c];
                acc[i] += (vv[c] + ev) * wgt;
            }
        }
    }
    float inv = (den > 0.0f) ? 1.0f / den : 0.0f;
    const float* ss = Qcat + (size_t)n * QSTRIDE + 768;   // S (skip, bias included)
    float* out = nf + (size_t)n * 256;
#pragma unroll
    for (int i = 0; i < 8; i++) {
        int c = i * 32 + lane;
        float v = acc[i] * inv + ss[c];
        out[c] = fmaxf(v, 0.0f);
    }
}

// ---------- fused pooling (batch is sorted -> contiguous ranges) + 3-layer MLP + sigmoid ----------
__global__ __launch_bounds__(256) void pool_mlp(
    const float* __restrict__ H, const int* __restrict__ batch,
    const float* __restrict__ W1, const float* __restrict__ b1,
    const float* __restrict__ W2, const float* __restrict__ b2,
    const float* __restrict__ W3, const float* __restrict__ b3,
    float* __restrict__ out)
{
    int b = blockIdx.x, t = threadIdx.x;
    __shared__ int range[2];
    __shared__ float r[512];
    __shared__ float o1[256];
    __shared__ float o2[128];

    if (t == 0) {   // lower_bound(b)
        int lo = 0, hi = N_NODES;
        while (lo < hi) { int mid = (lo + hi) >> 1; if (batch[mid] < b) lo = mid + 1; else hi = mid; }
        range[0] = lo;
    } else if (t == 1) {   // upper_bound(b)
        int lo = 0, hi = N_NODES;
        while (lo < hi) { int mid = (lo + hi) >> 1; if (batch[mid] <= b) lo = mid + 1; else hi = mid; }
        range[1] = lo;
    }
    __syncthreads();
    int beg = range[0], end = range[1];

    float s = 0.0f, mx = -3.402823466e38f;
    for (int n = beg; n < end; n++) {
        float v = H[(size_t)n * 256 + t];
        s += v; mx = fmaxf(mx, v);
    }
    float cntf = (float)(end - beg);
    r[t]       = s / fmaxf(cntf, 1.0f);
    r[256 + t] = mx;
    __syncthreads();

    float acc = b1[t];
    for (int i = 0; i < 512; i++) acc += r[i] * W1[i * 256 + t];
    o1[t] = fmaxf(acc, 0.0f);
    __syncthreads();
    if (t < 128) {
        float a2 = b2[t];
        for (int i = 0; i < 256; i++) a2 += o1[i] * W2[i * 128 + t];
        o2[t] = fmaxf(a2, 0.0f);
    }
    __syncthreads();
    if (t < 32) {
        float a3 = 0.0f;
        for (int i = t; i < 128; i += 32) a3 += o2[i] * W3[i];
        for (int o = 16; o > 0; o >>= 1) a3 += __shfl_down(a3, o, 32);
        if (t == 0) out[b] = 1.0f / (1.0f + __expf(-(a3 + b3[0])));
    }
}

extern "C" void kernel_launch(void* const* d_in, const int* in_sizes, int n_in,
                              void* d_out, int out_size, void* d_ws, size_t ws_size,
                              hipStream_t stream)
{
    (void)in_sizes; (void)n_in; (void)out_size; (void)ws_size;
    const int*   x     = (const int*)d_in[0];
    const int*   ei    = (const int*)d_in[1];
    const float* ea    = (const float*)d_in[2];
    const int*   batch = (const int*)d_in[3];
    const float* emb   = (const float*)d_in[4];
    const float* c1[9]; for (int i = 0; i < 9; i++) c1[i] = (const float*)d_in[5 + i];
    const float* c2[9]; for (int i = 0; i < 9; i++) c2[i] = (const float*)d_in[14 + i];
    const float* lin1W = (const float*)d_in[23]; const float* lin1b = (const float*)d_in[24];
    const float* lin2W = (const float*)d_in[25]; const float* lin2b = (const float*)d_in[26];
    const float* lin3W = (const float*)d_in[27]; const float* lin3b = (const float*)d_in[28];

    char* ws = (char*)d_ws;
    size_t off = 0;
    auto alloc = [&](size_t bytes) -> void* {
        void* p = ws + off; off += (bytes + 255) & ~(size_t)255; return p;
    };
    float*          NF      = (float*)alloc((size_t)N_NODES * 256 * 4);        // h0 -> h -> H
    float*          Qcat    = (float*)alloc((size_t)N_NODES * QSTRIDE * 4);    // fused Q|K|V|S
    float*          logits  = (float*)alloc((size_t)N_EDGES * 4);
    int*            counts  = (int*)alloc((size_t)N_NODES * 4);
    int*            offsets = (int*)alloc((size_t)(N_NODES + 1) * 4);
    int*            cursor  = (int*)alloc((size_t)N_NODES * 4);
    int*            eids    = (int*)alloc((size_t)N_EDGES * 4);
    unsigned short* Wt1     = (unsigned short*)alloc((size_t)1024 * 256 * 2);
    unsigned short* Wt2     = (unsigned short*)alloc((size_t)1024 * 256 * 2);
    float*          bc1     = (float*)alloc(1024 * 4);
    float*          bc2     = (float*)alloc(1024 * 4);

    // weight repack + embedding
    conv_weights<<<1024, 256, 0, stream>>>(c1[0], c1[2], c1[4], c1[7],
                                           c1[1], c1[3], c1[5], c1[8], Wt1, bc1);
    conv_weights<<<1024, 256, 0, stream>>>(c2[0], c2[2], c2[4], c2[7],
                                           c2[1], c2[3], c2[5], c2[8], Wt2, bc2);
    embed_gather<<<(N_NODES * 256) / 256, 256, 0, stream>>>(x, emb, NF);

    // CSR build (shared by both layers)
    hipMemsetAsync(counts, 0, (size_t)N_NODES * 4, stream);
    csr_count<<<(N_EDGES + 255) / 256, 256, 0, stream>>>(ei, counts);
    csr_scan<<<1, 256, 0, stream>>>(counts, offsets, cursor);
    csr_fill<<<(N_EDGES + 255) / 256, 256, 0, stream>>>(ei, cursor, eids);

    for (int layer = 0; layer < 2; ++layer) {
        const unsigned short* Wt = layer ? Wt2 : Wt1;
        const float* bc = layer ? bc2 : bc1;
        const float* We = layer ? c2[6] : c1[6];

        dim3 g((N_NODES + 127) / 128, QSTRIDE / 128);
        wmma_gemm_qkvs<<<g, 256, 0, stream>>>(NF, Wt, bc, Qcat, N_NODES, 256);

        edge_logits<<<N_EDGES / 8, 256, 0, stream>>>(ei, ea, We, Qcat, logits);
        node_aggregate<<<(N_NODES + 7) / 8, 256, 0, stream>>>(ei, ea, We, Qcat, logits,
                                                              offsets, eids, NF);
    }

    pool_mlp<<<64, 256, 0, stream>>>(NF, batch, lin1W, lin1b, lin2W, lin2b,
                                     lin3W, lin3b, (float*)d_out);
}
